// NeighborhoodAttention3D_9964324126858
// MI455X (gfx1250) — matrix operand order repaired and checked by reading, not compiled, not run on hardware
//
#include <hip/hip_runtime.h>

#define DEV __device__ __forceinline__

typedef __bf16 v8bf  __attribute__((ext_vector_type(8)));
typedef __bf16 v16bf __attribute__((ext_vector_type(16)));
typedef float  v8f   __attribute__((ext_vector_type(8)));

constexpr int Dd = 8, Hh = 24, Ww = 24;
constexpr int L   = Dd * Hh * Ww;   // 4608
constexpr int DIM = 192;
constexpr float QK_SCALE = 0.17677669529663687f; // 32^-0.5

// A-operand fragment (16x32, 16-bit): lane half h -> K elems {0..15}+16h
DEV v16bf load_frag_a(const __bf16* row, int half) {
  v8bf lo = *(const v8bf*)(row + 16 * half);
  v8bf hi = *(const v8bf*)(row + 16 * half + 8);
  v16bf r;
#pragma unroll
  for (int i = 0; i < 8; ++i) { r[i] = lo[i]; r[8 + i] = hi[i]; }
  return r;
}

// B-operand fragment (32x16, 16-bit) from row-major B^T (N x K): lane half h -> K elems {0..15}+16h
DEV v16bf load_frag_b(const __bf16* row, int half) {
  v8bf lo = *(const v8bf*)(row + 16 * half);
  v8bf hi = *(const v8bf*)(row + 16 * half + 8);
  v16bf r;
#pragma unroll
  for (int i = 0; i < 8; ++i) { r[i] = lo[i]; r[8 + i] = hi[i]; }
  return r;
}

DEV v8f wmma_bf16(v16bf a, v16bf b, v8f c) {
  return __builtin_amdgcn_wmma_f32_16x16x32_bf16(false, a, false, b, (short)0, c,
                                                 false, false);
}

// ---------------------------------------------------------------------------
// 1) fp32 -> bf16 convert; transpose weights to [N][K] for B-operand loads
// ---------------------------------------------------------------------------
__global__ __launch_bounds__(256) void convert_kernel(
    const float* __restrict__ x, const float* __restrict__ wqkv,
    const float* __restrict__ wproj, __bf16* __restrict__ xb,
    __bf16* __restrict__ wqt, __bf16* __restrict__ wpt) {
  int i = blockIdx.x * 256 + threadIdx.x;
  if (i < L * DIM) xb[i] = (__bf16)x[i];
  if (i < DIM * 3 * DIM) {
    int k = i / (3 * DIM), n = i % (3 * DIM);
    wqt[(size_t)n * DIM + k] = (__bf16)wqkv[i];
  }
  if (i < DIM * DIM) {
    int k = i / DIM, n = i % DIM;
    wpt[(size_t)n * DIM + k] = (__bf16)wproj[i];
  }
}

// ---------------------------------------------------------------------------
// 2) QKV GEMM: [4608x192] x [192x576] via WMMA bf16, one wave = 16x32 tile.
//    Epilogue: +bias, q pre-scaled, scatter to per-(scale,head) [pos][32].
// ---------------------------------------------------------------------------
__global__ __launch_bounds__(256) void qkv_gemm_kernel(
    const __bf16* __restrict__ xb, const __bf16* __restrict__ wqt,
    const float* __restrict__ bias, __bf16* __restrict__ qg,
    __bf16* __restrict__ kg, __bf16* __restrict__ vg) {
  const int lane = threadIdx.x & 31;
  const int l15 = lane & 15, half = lane >> 4;
  const int wid = (blockIdx.x * 256 + threadIdx.x) >> 5;  // 0..5183
  const int mt = wid / 18, nt = wid % 18;

  const __bf16* arow  = xb  + (size_t)(mt * 16 + l15) * DIM;
  const __bf16* brow0 = wqt + (size_t)(nt * 32 + l15) * DIM;
  const __bf16* brow1 = brow0 + (size_t)16 * DIM;

  v8f acc0 = {}, acc1 = {};
#pragma unroll
  for (int kk = 0; kk < 6; ++kk) {
    v16bf a  = load_frag_a(arow  + kk * 32, half);
    v16bf b0 = load_frag_b(brow0 + kk * 32, half);
    v16bf b1 = load_frag_b(brow1 + kk * 32, half);
    acc0 = wmma_bf16(a, b0, acc0);
    acc1 = wmma_bf16(a, b1, acc1);
  }

#pragma unroll
  for (int r = 0; r < 8; ++r) {
    int m = mt * 16 + r + 8 * half;  // position 0..4607
#pragma unroll
    for (int t = 0; t < 2; ++t) {
      int col = nt * 32 + t * 16 + l15;
      float v = (t ? acc1[r] : acc0[r]) + bias[col];
      int sc = col / DIM;
      int c2 = col - sc * DIM;
      int sel = c2 >> 6;            // 0=q 1=k 2=v
      int head = (c2 >> 5) & 1;
      int hd = c2 & 31;
      size_t off = ((size_t)(sc * 2 + head) * L + m) * 32 + hd;
      if (sel == 0)      qg[off] = (__bf16)(v * QK_SCALE);
      else if (sel == 1) kg[off] = (__bf16)v;
      else               vg[off] = (__bf16)v;
    }
  }
}

// ---------------------------------------------------------------------------
// 3) Neighborhood attention, flash-style over neighbor chunks of 32.
//    One wave = 16 queries along W at fixed (d,h) x 1 head.
// ---------------------------------------------------------------------------
template <int K>
__global__ __launch_bounds__(128) void natten_kernel(
    const __bf16* __restrict__ qg, const __bf16* __restrict__ kg,
    const __bf16* __restrict__ vg, __bf16* __restrict__ yb, int scale_idx) {
  constexpr int NW   = 16 + K - 1;        // union of windows along W for 16 queries
  constexpr int NTOT = K * K * NW;        // neighbors per query tile
  constexpr int NCH  = (NTOT + 31) / 32;  // 32-neighbor chunks

  __shared__ __align__(16) __bf16 Vlds[4][32][40];  // [wave][nbr][hd] (+pad)
  __shared__ __align__(16) __bf16 Plds[4][16][40];  // [wave][query][nbr] (+pad)

  const int lane  = threadIdx.x & 31;
  const int l15   = lane & 15;
  const int half  = lane >> 4;
  const int wslot = threadIdx.x >> 5;
  const int wid   = blockIdx.x * 4 + wslot;  // 0..767

  const int head = wid & 1;
  const int tile = (wid >> 1) & 1;
  const int hh   = (wid >> 2) % Hh;
  const int dd   = (wid >> 2) / Hh;
  const int w0   = tile * 16;

  const size_t hbase = (size_t)(scale_idx * 2 + head) * L * 32;
  const __bf16* qp = qg + hbase;
  const __bf16* kp = kg + hbase;
  const __bf16* vp = vg + hbase;

  const int dstart = min(max(dd - K / 2, 0), Dd - K);
  const int hstart = min(max(hh - K / 2, 0), Hh - K);
  const int wbase  = min(max(w0 - K / 2, 0), Ww - NW);

  int swr[8];  // per-row clamped window start along W
#pragma unroll
  for (int r = 0; r < 8; ++r) {
    int wqr = w0 + r + 8 * half;
    swr[r] = min(max(wqr - K / 2, 0), Ww - K);
  }

  // Q fragment (A operand): rows = queries; dummy rows clamp to w=23 (masked later)
  const int wq = min(w0 + l15, Ww - 1);
  const v16bf aQ = load_frag_a(qp + (size_t)((dd * Hh + hh) * Ww + wq) * 32, half);

  v8f acc0 = {}, acc1 = {};
  float rmax[8], rsum[8];
#pragma unroll
  for (int r = 0; r < 8; ++r) { rmax[r] = -1e30f; rsum[r] = 0.f; }

  for (int c = 0; c < NCH; ++c) {
    // ---- gather key chunk (B operand for logits; lane = neighbor column) ----
    int raw0 = c * 32 + l15;
    int raw1 = raw0 + 16;
    int nb0 = min(raw0, NTOT - 1);
    int nb1 = min(raw1, NTOT - 1);
    int wi0 = nb0 % NW, rest0 = nb0 / NW;
    int wi1 = nb1 % NW, rest1 = nb1 / NW;
    int wr0 = wbase + wi0, wr1 = wbase + wi1;
    int pos0 = ((dstart + rest0 / K) * Hh + (hstart + rest0 % K)) * Ww + min(wr0, Ww - 1);
    int pos1 = ((dstart + rest1 / K) * Hh + (hstart + rest1 % K)) * Ww + min(wr1, Ww - 1);
    v16bf bK0 = load_frag_b(kp + (size_t)pos0 * 32, half);
    v16bf bK1 = load_frag_b(kp + (size_t)pos1 * 32, half);

    // ---- stage V chunk into LDS: one lane per neighbor, 64B each ----
    int nbv = min(c * 32 + lane, NTOT - 1);
    int wiv = nbv % NW, restv = nbv / NW;
    int posv = ((dstart + restv / K) * Hh + (hstart + restv % K)) * Ww +
               min(wbase + wiv, Ww - 1);
    const v8bf* vsrc = (const v8bf*)(vp + (size_t)posv * 32);
#pragma unroll
    for (int j = 0; j < 4; ++j) *(v8bf*)&Vlds[wslot][lane][8 * j] = vsrc[j];

    // ---- logits: S[16q x 32nbr] = Q K^T ----
    v8f z = {};
    v8f s0 = wmma_bf16(aQ, bK0, z);
    v8f s1 = wmma_bf16(aQ, bK1, z);

    // ---- mask + online softmax (row reductions within 16-lane groups) ----
#pragma unroll
    for (int r = 0; r < 8; ++r) {
      bool v0 = (raw0 < NTOT) && (wr0 >= swr[r]) && (wr0 < swr[r] + K);
      bool v1 = (raw1 < NTOT) && (wr1 >= swr[r]) && (wr1 < swr[r] + K);
      float sv0 = v0 ? s0[r] : -1e30f;
      float sv1 = v1 ? s1[r] : -1e30f;
      float mx = fmaxf(sv0, sv1);
#pragma unroll
      for (int off = 1; off < 16; off <<= 1) mx = fmaxf(mx, __shfl_xor(mx, off, 32));
      float mnew = fmaxf(rmax[r], mx);
      float corr = __expf(rmax[r] - mnew);
      rmax[r] = mnew;
      float p0 = __expf(sv0 - mnew);
      float p1 = __expf(sv1 - mnew);
      float ps = p0 + p1;
#pragma unroll
      for (int off = 1; off < 16; off <<= 1) ps += __shfl_xor(ps, off, 32);
      rsum[r] = rsum[r] * corr + ps;
      acc0[r] *= corr;
      acc1[r] *= corr;
      Plds[wslot][r + 8 * half][l15]      = (__bf16)p0;
      Plds[wslot][r + 8 * half][16 + l15] = (__bf16)p1;
    }

    asm volatile("s_wait_dscnt 0" ::: "memory");

    // ---- AV: acc[16q x 32hd] += P V  (contraction over 32 chunk neighbors) ----
    v16bf aP = load_frag_a(&Plds[wslot][l15][0], half);
    v16bf bv0, bv1;
#pragma unroll
    for (int e = 0; e < 16; ++e) {
      int nbr = e + 16 * half;  // B-operand linear K map
      bv0[e] = Vlds[wslot][nbr][l15];
      bv1[e] = Vlds[wslot][nbr][16 + l15];
    }
    acc0 = wmma_bf16(aP, bv0, acc0);
    acc1 = wmma_bf16(aP, bv1, acc1);
  }

  // ---- epilogue: normalize rows, store valid queries to y (bf16) ----
  const int colbase = scale_idx * 64 + head * 32;
#pragma unroll
  for (int r = 0; r < 8; ++r) {
    int wqr = w0 + r + 8 * half;
    if (wqr < Ww) {
      float inv = rsum[r] > 0.f ? 1.f / rsum[r] : 0.f;
      size_t pos = (size_t)((dd * Hh + hh) * Ww + wqr);
      yb[pos * DIM + colbase + l15]      = (__bf16)(acc0[r] * inv);
      yb[pos * DIM + colbase + 16 + l15] = (__bf16)(acc1[r] * inv);
    }
  }
}

// ---------------------------------------------------------------------------
// 4) Output projection: [4608x192] x [192x192] + bias -> f32
// ---------------------------------------------------------------------------
__global__ __launch_bounds__(256) void proj_gemm_kernel(
    const __bf16* __restrict__ yb, const __bf16* __restrict__ wpt,
    const float* __restrict__ bias, float* __restrict__ out) {
  const int lane = threadIdx.x & 31;
  const int l15 = lane & 15, half = lane >> 4;
  const int wid = (blockIdx.x * 256 + threadIdx.x) >> 5;  // 0..1727
  const int mt = wid / 6, nt = wid % 6;

  const __bf16* arow  = yb  + (size_t)(mt * 16 + l15) * DIM;
  const __bf16* brow0 = wpt + (size_t)(nt * 32 + l15) * DIM;
  const __bf16* brow1 = brow0 + (size_t)16 * DIM;

  v8f acc0 = {}, acc1 = {};
#pragma unroll
  for (int kk = 0; kk < 6; ++kk) {
    v16bf a  = load_frag_a(arow  + kk * 32, half);
    v16bf b0 = load_frag_b(brow0 + kk * 32, half);
    v16bf b1 = load_frag_b(brow1 + kk * 32, half);
    acc0 = wmma_bf16(a, b0, acc0);
    acc1 = wmma_bf16(a, b1, acc1);
  }

#pragma unroll
  for (int r = 0; r < 8; ++r) {
    int m = mt * 16 + r + 8 * half;
#pragma unroll
    for (int t = 0; t < 2; ++t) {
      int col = nt * 32 + t * 16 + l15;
      out[(size_t)m * DIM + col] = (t ? acc1[r] : acc0[r]) + bias[col];
    }
  }
}

// ---------------------------------------------------------------------------
extern "C" void kernel_launch(void* const* d_in, const int* in_sizes, int n_in,
                              void* d_out, int out_size, void* d_ws, size_t ws_size,
                              hipStream_t stream) {
  (void)in_sizes; (void)n_in; (void)out_size; (void)ws_size;
  const float* x     = (const float*)d_in[0];
  const float* Wqkv  = (const float*)d_in[1];
  const float* bqkv  = (const float*)d_in[2];
  const float* Wproj = (const float*)d_in[3];
  const float* bproj = (const float*)d_in[4];
  float* out = (float*)d_out;

  char* ws = (char*)d_ws;
  auto take = [&](size_t bytes) {
    char* p = ws;
    ws += (bytes + 255) & ~(size_t)255;
    return p;
  };
  __bf16* xb  = (__bf16*)take((size_t)L * DIM * 2);
  __bf16* wqt = (__bf16*)take((size_t)3 * DIM * DIM * 2);
  __bf16* wpt = (__bf16*)take((size_t)DIM * DIM * 2);
  __bf16* qg  = (__bf16*)take((size_t)6 * L * 32 * 2);
  __bf16* kg  = (__bf16*)take((size_t)6 * L * 32 * 2);
  __bf16* vg  = (__bf16*)take((size_t)6 * L * 32 * 2);
  __bf16* yb  = (__bf16*)take((size_t)L * DIM * 2);

  convert_kernel<<<(L * DIM + 255) / 256, 256, 0, stream>>>(x, Wqkv, Wproj, xb, wqt, wpt);
  qkv_gemm_kernel<<<(288 * 18 * 32) / 256, 256, 0, stream>>>(xb, wqt, bqkv, qg, kg, vg);
  natten_kernel<3><<<192, 128, 0, stream>>>(qg, kg, vg, yb, 0);
  natten_kernel<5><<<192, 128, 0, stream>>>(qg, kg, vg, yb, 1);
  natten_kernel<7><<<192, 128, 0, stream>>>(qg, kg, vg, yb, 2);
  proj_gemm_kernel<<<(288 * 6 * 32) / 256, 256, 0, stream>>>(yb, wpt, bproj, out);
}
